// PermutEquivMP_81767587381702
// MI455X (gfx1250) — matrix operand order repaired and checked by reading, not compiled, not run on hardware
//
#include <hip/hip_runtime.h>

// GIN 2-layer forward for MI455X (gfx1250, wave32, WMMA + TDM).
// Per layer: zero(agg) -> scatter-add(x[src] -> agg[dst], HW f32 atomics) ->
// fused MLP (TDM weight staging, bf16x3 split-precision WMMA, h kept in LDS).

typedef __attribute__((ext_vector_type(16))) __bf16 v16bf;
typedef __attribute__((ext_vector_type(8)))  __bf16 v8bf;
typedef __attribute__((ext_vector_type(8)))  float  v8f;
typedef __attribute__((ext_vector_type(4)))  unsigned v4u;
typedef __attribute__((ext_vector_type(8)))  int    v8i;
typedef __attribute__((ext_vector_type(4)))  int    v4i;

#define WSTRIDE 136            // padded LDS row stride (elements) to spread banks
// LDS map: wa_hi 0 | wa_lo 34816 | wb_hi 69632 | wb_lo 104448 | h_hi 139264 |
//          h_lo 174080 | ba 208896 | bb 209408 | f32 stage 209920 (+65536)
#define STAGE_OFF 209920
#define SMEM_BYTES 275456      // <= 320KB WGP LDS

__device__ __forceinline__ v8f wmma_bf16(v16bf a, v16bf b, v8f c) {
  // (neg_a, A, neg_b, B, c_mod, C, reuse_a, reuse_b)
  return __builtin_amdgcn_wmma_f32_16x16x32_bf16(false, a, false, b, (short)0, c,
                                                 false, false);
}

// TDM: DMA a 128x128 f32 row-major tile from global into LDS (2D descriptor).
__device__ __forceinline__ void tdm_load_128x128_f32(const float* gsrc,
                                                     unsigned lds_off) {
  unsigned long long ga = (unsigned long long)(uintptr_t)gsrc;
  v4u g0;
  g0[0] = 1u;                                            // count=1 (valid), user
  g0[1] = lds_off;                                       // lds_addr (bytes)
  g0[2] = (unsigned)ga;                                  // global_addr[31:0]
  g0[3] = (unsigned)((ga >> 32) & 0x01FFFFFFull)         // global_addr[56:32]
        | 0x80000000u;                                   // type=2 ("image")
  v8i g1;
  g1[0] = 0x00020000;                                    // data_size=4B, mask=0
  g1[1] = (int)(128u << 16);                             // tensor_dim0 = 128
  g1[2] = (int)(128u << 16);                             // tensor_dim1 = 128
  g1[3] = (int)(128u << 16);                             // tile_dim0 = 128
  g1[4] = 128;                                           // tile_dim1 = 128
  g1[5] = 128;                                           // tensor_dim0_stride
  g1[6] = 0;
  g1[7] = 0;
  v4i z4 = {0, 0, 0, 0};                                 // 2D: groups 2/3 unused
  v8i z8 = {0, 0, 0, 0, 0, 0, 0, 0};
  // clang-23 / therock-10 form: (g0, g1, g2, g3, g_ext, cpol)
  __builtin_amdgcn_tensor_load_to_lds(g0, g1, z4, z4, z8, 0);
}

// ---------------------------------------------------------------- zero agg
__global__ void zero_kernel(float4* __restrict__ p, long n4) {
  long i = (long)blockIdx.x * blockDim.x + threadIdx.x;
  if (i < n4) p[i] = make_float4(0.f, 0.f, 0.f, 0.f);
}

// ------------------------------------------------------- edge scatter-add
// one wave per edge; lane l handles features [4l, 4l+3]
__global__ void scatter_kernel(const float* __restrict__ x,
                               const long long* __restrict__ ei,
                               float* __restrict__ agg, int E) {
  long t = (long)blockIdx.x * blockDim.x + threadIdx.x;
  long e = t >> 5;
  if (e >= E) return;
  int lane = (int)(t & 31);
  int src = (int)ei[e];
  int dst = (int)ei[(long)E + e];
  float4 v = *(const float4*)(x + (size_t)src * 128 + lane * 4);
  float* a = agg + (size_t)dst * 128 + lane * 4;
  unsafeAtomicAdd(a + 0, v.x);
  unsafeAtomicAdd(a + 1, v.y);
  unsafeAtomicAdd(a + 2, v.z);
  unsafeAtomicAdd(a + 3, v.w);
}

// ------------------------------------------------------------ fused MLP
// block = 256 threads = 8 waves; each wave computes 16 rows x 128 cols.
// y = relu((xin+agg) @ Wa^T + ba) @ Wb^T + bb  with bf16x3 WMMA.
__global__ __launch_bounds__(256)
void gin_mlp_kernel(const float* __restrict__ xin, const float* __restrict__ agg,
                    const float* __restrict__ wa, const float* __restrict__ ba,
                    const float* __restrict__ wb, const float* __restrict__ bb,
                    float* __restrict__ yout, int n) {
  extern __shared__ unsigned char smem[];
  __bf16* s_wa_hi = (__bf16*)(smem + 0);
  __bf16* s_wa_lo = (__bf16*)(smem + 34816);
  __bf16* s_wb_hi = (__bf16*)(smem + 69632);
  __bf16* s_wb_lo = (__bf16*)(smem + 104448);
  __bf16* s_h_hi  = (__bf16*)(smem + 139264);
  __bf16* s_h_lo  = (__bf16*)(smem + 174080);
  float*  s_ba    = (float*)(smem + 208896);
  float*  s_bb    = (float*)(smem + 209408);
  float*  s_stage = (float*)(smem + STAGE_OFF);

  const int tid = threadIdx.x;
  const int wv  = tid >> 5;
  const unsigned stage_lds = (unsigned)(uintptr_t)(void*)s_stage;

  // ---- TDM-stage Wa, then convert f32 -> bf16 hi/lo split panels in LDS ----
  if (wv == 0) {
    tdm_load_128x128_f32(wa, stage_lds);
    __builtin_amdgcn_s_wait_tensorcnt(0);
  }
  if (tid < 128) { s_ba[tid] = ba[tid]; s_bb[tid] = bb[tid]; }
  __syncthreads();
  for (int i = tid; i < 128 * 128; i += 256) {
    int nr = i >> 7, k = i & 127;
    float w  = s_stage[i];
    __bf16 h = (__bf16)w;
    s_wa_hi[nr * WSTRIDE + k] = h;
    s_wa_lo[nr * WSTRIDE + k] = (__bf16)(w - (float)h);
  }
  __syncthreads();
  // ---- TDM-stage Wb (reuse staging buffer), convert ----
  if (wv == 0) {
    tdm_load_128x128_f32(wb, stage_lds);
    __builtin_amdgcn_s_wait_tensorcnt(0);
  }
  __syncthreads();
  for (int i = tid; i < 128 * 128; i += 256) {
    int nr = i >> 7, k = i & 127;
    float w  = s_stage[i];
    __bf16 h = (__bf16)w;
    s_wb_hi[nr * WSTRIDE + k] = h;
    s_wb_lo[nr * WSTRIDE + k] = (__bf16)(w - (float)h);
  }
  __syncthreads();

  const int lane = tid & 31;
  const int kgrp = lane >> 4;   // lane group (0: lanes 0-15, 1: lanes 16-31)
  const int rsel = lane & 15;

  const int  mrow0 = wv * 16;                                  // wave tile row
  const long grow  = (long)blockIdx.x * 128 + mrow0 + rsel;    // A-row (global)
  const bool valid = grow < n;
  const float* xr = xin + (size_t)(valid ? grow : 0) * 128;
  const float* ar = agg + (size_t)(valid ? grow : 0) * 128;

  v8f acc[8];

  // ============================ GEMM1: (x+agg) @ Wa^T + ba, relu ============
  #pragma unroll
  for (int nt = 0; nt < 8; ++nt) {
    float bv = s_ba[nt * 16 + rsel];
    #pragma unroll
    for (int r = 0; r < 8; ++r) acc[nt][r] = bv;
  }

  #pragma unroll
  for (int kc = 0; kc < 4; ++kc) {
    // A fragment: 16-bit A 16x32 layout -> per lane K in {b..b+7, b+16..b+23}
    const int base = kc * 32 + kgrp * 8;
    float f[16];
    if (valid) {
      float4 x0 = *(const float4*)(xr + base);
      float4 x1 = *(const float4*)(xr + base + 4);
      float4 x2 = *(const float4*)(xr + base + 16);
      float4 x3 = *(const float4*)(xr + base + 20);
      float4 g0 = *(const float4*)(ar + base);
      float4 g1 = *(const float4*)(ar + base + 4);
      float4 g2 = *(const float4*)(ar + base + 16);
      float4 g3 = *(const float4*)(ar + base + 20);
      f[0]=x0.x+g0.x; f[1]=x0.y+g0.y; f[2]=x0.z+g0.z; f[3]=x0.w+g0.w;
      f[4]=x1.x+g1.x; f[5]=x1.y+g1.y; f[6]=x1.z+g1.z; f[7]=x1.w+g1.w;
      f[8]=x2.x+g2.x; f[9]=x2.y+g2.y; f[10]=x2.z+g2.z; f[11]=x2.w+g2.w;
      f[12]=x3.x+g3.x; f[13]=x3.y+g3.y; f[14]=x3.z+g3.z; f[15]=x3.w+g3.w;
    } else {
      #pragma unroll
      for (int j = 0; j < 16; ++j) f[j] = 0.f;
    }
    v16bf a_hi, a_lo;
    #pragma unroll
    for (int j = 0; j < 16; ++j) {
      __bf16 h = (__bf16)f[j];
      a_hi[j] = h;
      a_lo[j] = (__bf16)(f[j] - (float)h);
    }

    #pragma unroll
    for (int nt = 0; nt < 8; ++nt) {
      // B fragment: lane = column n, K in {b..b+7, b+16..b+23} per lane group
      const __bf16* ph = s_wa_hi + (nt * 16 + rsel) * WSTRIDE + kc * 32 + kgrp * 8;
      const __bf16* pl = s_wa_lo + (nt * 16 + rsel) * WSTRIDE + kc * 32 + kgrp * 8;
      v8bf b0 = *(const v8bf*)ph, b1 = *(const v8bf*)(ph + 16);
      v8bf c0 = *(const v8bf*)pl, c1 = *(const v8bf*)(pl + 16);
      v16bf b_hi, b_lo;
      #pragma unroll
      for (int j = 0; j < 8; ++j) {
        b_hi[j] = b0[j]; b_hi[j + 8] = b1[j];
        b_lo[j] = c0[j]; b_lo[j + 8] = c1[j];
      }
      acc[nt] = wmma_bf16(a_hi, b_hi, acc[nt]);
      acc[nt] = wmma_bf16(a_hi, b_lo, acc[nt]);
      acc[nt] = wmma_bf16(a_lo, b_hi, acc[nt]);
    }
  }

  // relu, split to bf16 hi/lo, stash h in LDS (C/D layout: VGPR r -> M=r+8*kgrp)
  #pragma unroll
  for (int nt = 0; nt < 8; ++nt) {
    #pragma unroll
    for (int r = 0; r < 8; ++r) {
      float v = acc[nt][r];
      v = v > 0.f ? v : 0.f;
      int mr  = mrow0 + kgrp * 8 + r;
      int col = nt * 16 + rsel;
      __bf16 h = (__bf16)v;
      s_h_hi[mr * WSTRIDE + col] = h;
      s_h_lo[mr * WSTRIDE + col] = (__bf16)(v - (float)h);
    }
  }
  __syncthreads();

  // ============================ GEMM2: h @ Wb^T + bb ========================
  #pragma unroll
  for (int nt = 0; nt < 8; ++nt) {
    float bv = s_bb[nt * 16 + rsel];
    #pragma unroll
    for (int r = 0; r < 8; ++r) acc[nt][r] = bv;
  }

  const int arow = mrow0 + rsel;
  #pragma unroll
  for (int kc = 0; kc < 4; ++kc) {
    const int base = kc * 32 + kgrp * 8;
    const __bf16* ph = s_h_hi + arow * WSTRIDE + base;
    const __bf16* pl = s_h_lo + arow * WSTRIDE + base;
    v8bf h0 = *(const v8bf*)ph, h1 = *(const v8bf*)(ph + 16);
    v8bf l0 = *(const v8bf*)pl, l1 = *(const v8bf*)(pl + 16);
    v16bf a_hi, a_lo;
    #pragma unroll
    for (int j = 0; j < 8; ++j) {
      a_hi[j] = h0[j]; a_hi[j + 8] = h1[j];
      a_lo[j] = l0[j]; a_lo[j + 8] = l1[j];
    }
    #pragma unroll
    for (int nt = 0; nt < 8; ++nt) {
      const __bf16* bh = s_wb_hi + (nt * 16 + rsel) * WSTRIDE + kc * 32 + kgrp * 8;
      const __bf16* bl = s_wb_lo + (nt * 16 + rsel) * WSTRIDE + kc * 32 + kgrp * 8;
      v8bf b0 = *(const v8bf*)bh, b1 = *(const v8bf*)(bh + 16);
      v8bf c0 = *(const v8bf*)bl, c1 = *(const v8bf*)(bl + 16);
      v16bf b_hi, b_lo;
      #pragma unroll
      for (int j = 0; j < 8; ++j) {
        b_hi[j] = b0[j]; b_hi[j + 8] = b1[j];
        b_lo[j] = c0[j]; b_lo[j + 8] = c1[j];
      }
      acc[nt] = wmma_bf16(a_hi, b_hi, acc[nt]);
      acc[nt] = wmma_bf16(a_hi, b_lo, acc[nt]);
      acc[nt] = wmma_bf16(a_lo, b_hi, acc[nt]);
    }
  }

  // store y (f32), rows guarded against N
  #pragma unroll
  for (int nt = 0; nt < 8; ++nt) {
    #pragma unroll
    for (int r = 0; r < 8; ++r) {
      long row = (long)blockIdx.x * 128 + mrow0 + kgrp * 8 + r;
      if (row < n) yout[row * 128 + nt * 16 + rsel] = acc[nt][r];
    }
  }
}

// ------------------------------------------------------------------ launch
extern "C" void kernel_launch(void* const* d_in, const int* in_sizes, int n_in,
                              void* d_out, int out_size, void* d_ws, size_t ws_size,
                              hipStream_t stream) {
  const float*     x   = (const float*)d_in[0];
  const long long* ei  = (const long long*)d_in[1];   // int64 edge_index [2,E]
  const float* w0a = (const float*)d_in[2];
  const float* b0a = (const float*)d_in[3];
  const float* w0b = (const float*)d_in[4];
  const float* b0b = (const float*)d_in[5];
  const float* w1a = (const float*)d_in[6];
  const float* b1a = (const float*)d_in[7];
  const float* w1b = (const float*)d_in[8];
  const float* b1b = (const float*)d_in[9];
  float* out = (float*)d_out;

  const int N = in_sizes[0] / 128;
  const int E = in_sizes[1] / 2;

  float* agg  = (float*)d_ws;                 // N*128 f32
  float* hmid = agg + (size_t)N * 128;        // N*128 f32

  (void)hipFuncSetAttribute(reinterpret_cast<const void*>(gin_mlp_kernel),
                            hipFuncAttributeMaxDynamicSharedMemorySize, SMEM_BYTES);

  const long n4 = (long)N * 128 / 4;
  const int  zb = (int)((n4 + 255) / 256);
  const long st = (long)E * 32;
  const int  sb = (int)((st + 255) / 256);
  const int  mb = (N + 127) / 128;

  // ---- layer 1
  zero_kernel<<<zb, 256, 0, stream>>>((float4*)agg, n4);
  scatter_kernel<<<sb, 256, 0, stream>>>(x, ei, agg, E);
  gin_mlp_kernel<<<mb, 256, SMEM_BYTES, stream>>>(x, agg, w0a, b0a, w0b, b0b, hmid, N);
  // ---- layer 2
  zero_kernel<<<zb, 256, 0, stream>>>((float4*)agg, n4);
  scatter_kernel<<<sb, 256, 0, stream>>>(hmid, ei, agg, E);
  gin_mlp_kernel<<<mb, 256, SMEM_BYTES, stream>>>(hmid, agg, w1a, b1a, w1b, b1b, out, N);
}
